// caponly_81312320848615
// MI455X (gfx1250) — compile-verified, matrix operand order checked by reading, not run on hardware
//
#include <hip/hip_runtime.h>
#include <math.h>

// Problem constants (match reference)
#define NG     8
#define LSEQ   4096          // H*W = 64*64
#define CDIM   256
#define NSTATE 8
#define RNK    16
#define KCONV  4
#define DEPTH  4

typedef __attribute__((ext_vector_type(16))) _Float16 v16h;
typedef __attribute__((ext_vector_type(8)))  float    v8f;
typedef unsigned int u32x4 __attribute__((ext_vector_type(4)));
typedef int          i32x4 __attribute__((ext_vector_type(4)));
typedef int          i32x8 __attribute__((ext_vector_type(8)));

union HFrag { v16h v; _Float16 h[16]; };
union U4  { u32x4 v; unsigned int e[4]; };
union I8  { i32x8 v; int e[8]; };

// ---------------------------------------------------------------------------
// Tensor Data Mover: async 2D tile load Global -> LDS (f32 elements).
// Builds D# group0/group1 per CDNA5 ISA.
//   tile_d0  : contiguous elements per line (dim0)
//   tile_d1  : number of lines (dim1)
//   stride0  : tensor dim0 stride (elements) between lines
// Issue once per wave (EXEC ignored by TDM); caller must s_wait_tensorcnt.
// ---------------------------------------------------------------------------
__device__ inline void tdm_load_2d_f32(unsigned ldsOff, const float* gaddr,
                                       unsigned tensor_d0, unsigned tensor_d1,
                                       unsigned tile_d0, unsigned tile_d1,
                                       unsigned stride0) {
  unsigned long long ga = (unsigned long long)(uintptr_t)gaddr;
  U4 g0;
  g0.e[0] = 1u;                                   // count=1, user descriptor
  g0.e[1] = ldsOff;                               // lds_addr (bytes)
  g0.e[2] = (unsigned)(ga & 0xFFFFFFFFu);         // global_addr[31:0]
  g0.e[3] = (unsigned)((ga >> 32) & 0x1FFFFFFu)   // global_addr[56:32]
          | (2u << 30);                           // type = 2 ("image")
  I8 g1;
  g1.e[0] = (int)(2u << 16);                      // data_size = 2 -> 4 bytes
  g1.e[1] = (int)((tensor_d0 & 0xFFFFu) << 16);   // tensor_dim0[15:0]
  g1.e[2] = (int)(((tensor_d0 >> 16) & 0xFFFFu)
          | ((tensor_d1 & 0xFFFFu) << 16));       // dim0[31:16] | dim1[15:0]
  g1.e[3] = (int)(((tensor_d1 >> 16) & 0xFFFFu)
          | ((tile_d0 & 0xFFFFu) << 16));         // dim1[31:16] | tile_dim0
  g1.e[4] = (int)(tile_d1 & 0xFFFFu);             // tile_dim1 (tile_dim2=0)
  g1.e[5] = (int)stride0;                         // tensor_dim0_stride[31:0]
  g1.e[6] = 0;
  g1.e[7] = 0;
  i32x4 z4 = {0, 0, 0, 0};
#if __clang_major__ >= 23
  i32x8 z8 = {0, 0, 0, 0, 0, 0, 0, 0};
  __builtin_amdgcn_tensor_load_to_lds(g0.v, g1.v, z4, z4, z8, 0);
#else
  __builtin_amdgcn_tensor_load_to_lds(g0.v, g1.v, z4, z4, 0);
#endif
}

// ---------------------------------------------------------------------------
// 1) Segment mean over agents: data (sum(rl), C, L) -> x (G, L, C)
// ---------------------------------------------------------------------------
__global__ __launch_bounds__(256) void k_segmean(const float* __restrict__ data,
                                                 const int* __restrict__ rl,
                                                 float* __restrict__ x) {
  int idx = blockIdx.x * 256 + threadIdx.x;           // over NG*LSEQ*CDIM
  int c = idx % CDIM;
  int r = idx / CDIM;
  int l = r % LSEQ;
  int g = r / LSEQ;
  int base = 0;
  for (int q = 0; q < NG; ++q) if (q < g) base += rl[q];
  int cnt = rl[g];
  float s = 0.f;
  for (int a = 0; a < cnt; ++a)
    s += data[((size_t)((base + a) * CDIM + c)) * LSEQ + l];
  x[((size_t)(g * LSEQ + l)) * CDIM + c] = s / (float)cnt;
}

// ---------------------------------------------------------------------------
// 2) LayerNorm over C=256: 8 rows per 256-thread block, 32 lanes per row
// ---------------------------------------------------------------------------
__global__ __launch_bounds__(256) void k_layernorm(const float* __restrict__ x,
                                                   const float* __restrict__ w,
                                                   const float* __restrict__ b,
                                                   float* __restrict__ xn) {
  int row  = blockIdx.x * 8 + (threadIdx.x >> 5);
  int lane = threadIdx.x & 31;
  const float* rp = x + (size_t)row * CDIM;
  float v[8], s = 0.f, ss = 0.f;
  #pragma unroll
  for (int q = 0; q < 8; ++q) {
    v[q] = rp[lane + q * 32];
    s += v[q]; ss += v[q] * v[q];
  }
  #pragma unroll
  for (int off = 16; off >= 1; off >>= 1) {
    s  += __shfl_xor(s,  off, 32);
    ss += __shfl_xor(ss, off, 32);
  }
  float mean = s * (1.f / CDIM);
  float var  = ss * (1.f / CDIM) - mean * mean;
  float rs   = rsqrtf(var + 1e-5f);
  #pragma unroll
  for (int q = 0; q < 8; ++q) {
    int c = lane + q * 32;
    xn[(size_t)row * CDIM + c] = (v[q] - mean) * rs * w[c] + b[c];
  }
}

// ---------------------------------------------------------------------------
// 3) in_proj GEMM: xz(M,512) = xn(M,256) x W(512,256)^T, f16 WMMA 16x16x32
//    Block = 16 rows x 128 cols, 8 waves (1 col-tile each), LDS-staged tiles.
//    K loop fully unrolled -> 8 back-to-back WMMA issues per wave.
// ---------------------------------------------------------------------------
__global__ __launch_bounds__(256) void k_gemm_inproj(const float* __restrict__ A,
                                                     const float* __restrict__ Bw,
                                                     float* __restrict__ Cout) {
  __shared__ _Float16 lA[16 * 32];
  __shared__ _Float16 lB[128 * 32];
  const int tid  = threadIdx.x;
  const int wave = tid >> 5;
  const int lane = tid & 31;
  const int m0   = blockIdx.x * 16;
  const int n0   = blockIdx.y * 128;
  v8f acc = {};
  #pragma unroll
  for (int k0 = 0; k0 < CDIM; k0 += 32) {
    #pragma unroll
    for (int e = 0; e < 2; ++e) {               // 16x32 A tile
      int i = tid + e * 256;
      int rr = i >> 5, kk = i & 31;
      lA[i] = (_Float16)A[(size_t)(m0 + rr) * CDIM + k0 + kk];
    }
    #pragma unroll
    for (int e = 0; e < 16; ++e) {              // 128x32 B tile
      int i = tid + e * 256;
      int cc = i >> 5, kk = i & 31;
      lB[i] = (_Float16)Bw[(size_t)(n0 + cc) * CDIM + k0 + kk];
    }
    if (k0 + 32 < CDIM)
      __builtin_prefetch(&Bw[(size_t)n0 * CDIM + k0 + 32], 0, 1);
    __syncthreads();
    HFrag fa, fb;
    const int row = lane & 15, hi = lane >> 4;
    #pragma unroll
    for (int v = 0; v < 8; ++v)
      #pragma unroll
      for (int p = 0; p < 2; ++p) {
        int kk = ((v & 4) ? 16 : 0) + hi * 8 + (v & 3) * 2 + p;
        fa.h[v * 2 + p] = lA[row * 32 + kk];
        fb.h[v * 2 + p] = lB[(wave * 16 + row) * 32 + kk];
      }
    acc = __builtin_amdgcn_wmma_f32_16x16x32_f16(false, fa.v, false, fb.v,
                                                 (short)0, acc, false, false);
    __syncthreads();
  }
  const int col = lane & 15, hi = lane >> 4;
  #pragma unroll
  for (int r = 0; r < 8; ++r)
    Cout[(size_t)(m0 + r + hi * 8) * 512 + n0 + wave * 16 + col] = acc[r];
}

// ---------------------------------------------------------------------------
// 4) Depthwise causal conv (k=4) + SiLU; reads xx-half of xz in scan order
// ---------------------------------------------------------------------------
__global__ __launch_bounds__(256) void k_conv_silu(const float* __restrict__ xz,
                                                   const float* __restrict__ cw,
                                                   const float* __restrict__ cb,
                                                   float* __restrict__ xc, int dir) {
  int idx = blockIdx.x * 256 + threadIdx.x;   // NG*CDIM*LSEQ
  int t = idx & (LSEQ - 1);
  int r = idx >> 12;                          // g*256 + d
  int d = r & 255;
  int g = r >> 8;
  float acc = cb[d];
  #pragma unroll
  for (int j = 0; j < KCONV; ++j) {
    int s = t - (KCONV - 1) + j;
    if (s >= 0) {
      int lpos = dir ? (LSEQ - 1 - s) : s;
      acc += cw[d * KCONV + j] * xz[((size_t)(g * LSEQ + lpos)) * 512 + d];
    }
  }
  float sig = 1.f / (1.f + __expf(-acc));
  xc[(size_t)r * LSEQ + t] = acc * sig;
}

// ---------------------------------------------------------------------------
// 5) x_proj: dtBC (G,32,L) = x_proj_w(32,256) x xc(G,256,L)
// ---------------------------------------------------------------------------
__global__ __launch_bounds__(256) void k_xproj(const float* __restrict__ xc,
                                               const float* __restrict__ xpw,
                                               float* __restrict__ dtBC) {
  int idx = blockIdx.x * 256 + threadIdx.x;   // NG*32*LSEQ
  int t = idx & (LSEQ - 1);
  int r = idx >> 12;                          // g*32 + e
  int e = r & 31;
  int g = r >> 5;
  float s = 0.f;
  for (int d = 0; d < CDIM; ++d)
    s += xpw[e * CDIM + d] * xc[((size_t)(g * CDIM + d)) * LSEQ + t];
  dtBC[(size_t)r * LSEQ + t] = s;
}

// ---------------------------------------------------------------------------
// 6) delta = softplus(dt_proj_w x dt + dt_proj_b)
// ---------------------------------------------------------------------------
__global__ __launch_bounds__(256) void k_delta(const float* __restrict__ dtBC,
                                               const float* __restrict__ dtw,
                                               const float* __restrict__ dtb,
                                               float* __restrict__ delta) {
  int idx = blockIdx.x * 256 + threadIdx.x;   // NG*CDIM*LSEQ
  int t = idx & (LSEQ - 1);
  int r = idx >> 12;
  int d = r & 255;
  int g = r >> 8;
  float s = dtb[d];
  #pragma unroll
  for (int rr = 0; rr < RNK; ++rr)
    s += dtw[d * RNK + rr] * dtBC[((size_t)(g * 32 + rr)) * LSEQ + t];
  float sp = (s > 20.f) ? s : log1pf(__expf(s));
  delta[(size_t)r * LSEQ + t] = sp;
}

// ---------------------------------------------------------------------------
// 7) Selective scan: 8 lanes per (g,d) channel (one per state n), sequential L
//    y = (scan + D*u) * silu(z), accumulated 0.5*(fwd+bwd) into ybuf (G,D,L)
// ---------------------------------------------------------------------------
__global__ __launch_bounds__(256) void k_scan(const float* __restrict__ xc,
                                              const float* __restrict__ delta,
                                              const float* __restrict__ dtBC,
                                              const float* __restrict__ xz,
                                              const float* __restrict__ Alog,
                                              const float* __restrict__ Dp,
                                              float* __restrict__ ybuf, int dir) {
  int tid = threadIdx.x;
  int n   = tid & 7;
  int grp = tid >> 3;
  int ch  = blockIdx.x * 32 + grp;            // g*256 + d
  int g   = ch >> 8;
  int d   = ch & 255;
  float A  = -__expf(Alog[d * NSTATE + n]);
  float Dd = Dp[d];
  float h  = 0.f;
  const float* dch  = delta + (size_t)ch * LSEQ;
  const float* uch  = xc    + (size_t)ch * LSEQ;
  const float* Brow = dtBC  + (size_t)(g * 32 + RNK + n) * LSEQ;
  const float* Crow = dtBC  + (size_t)(g * 32 + RNK + NSTATE + n) * LSEQ;
  for (int t = 0; t < LSEQ; ++t) {
    float dt = dch[t];
    float u  = uch[t];
    h = __expf(dt * A) * h + (dt * u) * Brow[t];
    float p = h * Crow[t];
    p += __shfl_xor(p, 4, 8);
    p += __shfl_xor(p, 2, 8);
    p += __shfl_xor(p, 1, 8);
    if (n == 0) {
      float y = p + Dd * u;
      int lpos = dir ? (LSEQ - 1 - t) : t;
      float z  = xz[((size_t)(g * LSEQ + lpos)) * 512 + CDIM + d];
      float sil = z / (1.f + __expf(-z));
      float val = 0.5f * y * sil;
      size_t oi = (size_t)ch * LSEQ + lpos;
      if (dir) ybuf[oi] += val; else ybuf[oi] = val;
    }
  }
}

// ---------------------------------------------------------------------------
// 8) out_proj GEMM (WMMA, TDM-fed tiles) + mean-norm LN + residual, fused.
//    Block = 16 rows x full 256 cols (8 waves x 2 col-tiles).
//    A tile  (16 l contiguous x 32 d, stride L)  via tensor_load_to_lds
//    B tile  (32 k contiguous x 256 rows, stride 256) via tensor_load_to_lds
// ---------------------------------------------------------------------------
__global__ __launch_bounds__(256) void k_gemm_outproj_ln(const float* __restrict__ Y,
                                                         const float* __restrict__ Bw,
                                                         const float* __restrict__ mnw,
                                                         const float* __restrict__ mnb,
                                                         float* __restrict__ x) {
  __shared__ float lAf[32 * 16];    // [d-line][l] after TDM (dim0 = 16 l's)
  __shared__ float lBf[256 * 32];   // [row][k] after TDM   (dim0 = 32 k's)
  __shared__ float lO[16 * 256];
  const int tid  = threadIdx.x;
  const int wave = tid >> 5;
  const int lane = tid & 31;
  const int m0   = blockIdx.x * 16;
  const int g    = m0 / LSEQ;
  const int l0   = m0 & (LSEQ - 1);
  const unsigned ldsA = (unsigned)(uintptr_t)(&lAf[0]);
  const unsigned ldsB = (unsigned)(uintptr_t)(&lBf[0]);
  v8f acc0 = {}, acc1 = {};
  #pragma unroll
  for (int k0 = 0; k0 < CDIM; k0 += 32) {
    if (tid < 32) {   // one TDM issue per block (wave 0); EXEC ignored by TDM
      // A tile: rows = 32 d-lines (stride L), line = 16 contiguous l's
      tdm_load_2d_f32(ldsA, &Y[((size_t)(g * CDIM + k0)) * LSEQ + l0],
                      /*tensor_d0=*/LSEQ, /*tensor_d1=*/CDIM,
                      /*tile_d0=*/16, /*tile_d1=*/32, /*stride0=*/LSEQ);
      // B tile: 256 rows (stride 256), line = 32 contiguous k's
      tdm_load_2d_f32(ldsB, &Bw[k0],
                      /*tensor_d0=*/CDIM, /*tensor_d1=*/CDIM,
                      /*tile_d0=*/32, /*tile_d1=*/CDIM, /*stride0=*/CDIM);
      __builtin_amdgcn_s_wait_tensorcnt(0);
    }
    __syncthreads();
    HFrag fa, fb0, fb1;
    const int row = lane & 15, hi = lane >> 4;
    #pragma unroll
    for (int v = 0; v < 8; ++v)
      #pragma unroll
      for (int p = 0; p < 2; ++p) {
        int kk = ((v & 4) ? 16 : 0) + hi * 8 + (v & 3) * 2 + p;
        fa.h[v * 2 + p]  = (_Float16)lAf[kk * 16 + row];
        fb0.h[v * 2 + p] = (_Float16)lBf[((wave * 2 + 0) * 16 + row) * 32 + kk];
        fb1.h[v * 2 + p] = (_Float16)lBf[((wave * 2 + 1) * 16 + row) * 32 + kk];
      }
    acc0 = __builtin_amdgcn_wmma_f32_16x16x32_f16(false, fa.v, false, fb0.v,
                                                  (short)0, acc0, false, false);
    acc1 = __builtin_amdgcn_wmma_f32_16x16x32_f16(false, fa.v, false, fb1.v,
                                                  (short)0, acc1, false, false);
    __syncthreads();
  }
  {
    const int col = lane & 15, hi = lane >> 4;
    #pragma unroll
    for (int r = 0; r < 8; ++r) {
      lO[(r + hi * 8) * 256 + wave * 32 + col]      = acc0[r];
      lO[(r + hi * 8) * 256 + wave * 32 + 16 + col] = acc1[r];
    }
  }
  __syncthreads();
  // mean-norm over 256 + residual: 16 threads per row
  const int row2 = tid >> 4, sub = tid & 15;
  float vals[16], s = 0.f, ss = 0.f;
  #pragma unroll
  for (int q = 0; q < 16; ++q) {
    vals[q] = lO[row2 * 256 + sub + q * 16];
    s += vals[q]; ss += vals[q] * vals[q];
  }
  #pragma unroll
  for (int off = 8; off >= 1; off >>= 1) {
    s  += __shfl_xor(s,  off, 16);
    ss += __shfl_xor(ss, off, 16);
  }
  float mean = s * (1.f / CDIM);
  float var  = ss * (1.f / CDIM) - mean * mean;
  float rs   = rsqrtf(var + 1e-5f);
  #pragma unroll
  for (int q = 0; q < 16; ++q) {
    int c = sub + q * 16;
    size_t gi = (size_t)(m0 + row2) * CDIM + c;
    x[gi] = (vals[q] - mean) * rs * mnw[c] + mnb[c] + x[gi];
  }
}

// ---------------------------------------------------------------------------
// 9) Final transpose x (G,L,C) -> out (G,C,H,W)
// ---------------------------------------------------------------------------
__global__ __launch_bounds__(256) void k_out(const float* __restrict__ x,
                                             float* __restrict__ out) {
  int idx = blockIdx.x * 256 + threadIdx.x;   // (g*256+c)*4096 + l
  int l = idx & (LSEQ - 1);
  int r = idx >> 12;
  int c = r & 255;
  int g = r >> 8;
  out[idx] = x[((size_t)(g * LSEQ + l)) * CDIM + c];
}

// ---------------------------------------------------------------------------
extern "C" void kernel_launch(void* const* d_in, const int* in_sizes, int n_in,
                              void* d_out, int out_size, void* d_ws, size_t ws_size,
                              hipStream_t stream) {
  (void)in_sizes; (void)n_in; (void)out_size; (void)ws_size;
  const float* data   = (const float*)d_in[0];
  const float* ln_w   = (const float*)d_in[1];
  const float* ln_b   = (const float*)d_in[2];
  const float* ipw    = (const float*)d_in[3];
  const float* conv_w = (const float*)d_in[4];
  const float* conv_b = (const float*)d_in[5];
  const float* xpw    = (const float*)d_in[6];
  const float* dtw    = (const float*)d_in[7];
  const float* dtb    = (const float*)d_in[8];
  const float* A_log  = (const float*)d_in[9];
  const float* Ab_log = (const float*)d_in[10];
  const float* Dp     = (const float*)d_in[11];
  const float* opw    = (const float*)d_in[12];
  const float* mn_w   = (const float*)d_in[13];
  const float* mn_b   = (const float*)d_in[14];
  const int*   rl     = (const int*)d_in[15];

  const size_t XSZ = (size_t)NG * LSEQ * CDIM;      // 8M floats
  float* W     = (float*)d_ws;
  float* x     = W;
  float* xz    = x + XSZ;
  float* xc    = xz + 2 * XSZ;
  float* dtBC  = xc + XSZ;
  float* delta = dtBC + (size_t)NG * 32 * LSEQ;
  float* ybuf  = delta + XSZ;
  float* xn    = ybuf + XSZ;

  k_segmean<<<(int)(XSZ / 256), 256, 0, stream>>>(data, rl, x);

  for (int layer = 0; layer < DEPTH; ++layer) {
    const float* lw  = ln_w   + (size_t)layer * CDIM;
    const float* lb  = ln_b   + (size_t)layer * CDIM;
    const float* ip  = ipw    + (size_t)layer * 512 * CDIM;
    const float* cw  = conv_w + (size_t)layer * CDIM * KCONV;
    const float* cb  = conv_b + (size_t)layer * CDIM;
    const float* xp  = xpw    + (size_t)layer * 32 * CDIM;
    const float* dw  = dtw    + (size_t)layer * CDIM * RNK;
    const float* db  = dtb    + (size_t)layer * CDIM;
    const float* Dl  = Dp     + (size_t)layer * CDIM;
    const float* op  = opw    + (size_t)layer * CDIM * CDIM;
    const float* mw  = mn_w   + (size_t)layer * CDIM;
    const float* mb  = mn_b   + (size_t)layer * CDIM;

    k_layernorm<<<NG * LSEQ / 8, 256, 0, stream>>>(x, lw, lb, xn);

    dim3 gIn(NG * LSEQ / 16, 512 / 128);
    k_gemm_inproj<<<gIn, 256, 0, stream>>>(xn, ip, xz);

    for (int dir = 0; dir < 2; ++dir) {
      const float* Al = (dir ? Ab_log : A_log) + (size_t)layer * CDIM * NSTATE;
      k_conv_silu<<<(int)(XSZ / 256), 256, 0, stream>>>(xz, cw, cb, xc, dir);
      k_xproj<<<NG * 32 * LSEQ / 256, 256, 0, stream>>>(xc, xp, dtBC);
      k_delta<<<(int)(XSZ / 256), 256, 0, stream>>>(dtBC, dw, db, delta);
      k_scan<<<NG * CDIM / 32, 256, 0, stream>>>(xc, delta, dtBC, xz, Al, Dl, ybuf, dir);
    }

    k_gemm_outproj_ln<<<NG * LSEQ / 16, 256, 0, stream>>>(ybuf, op, mw, mb, x);
  }

  k_out<<<(int)(XSZ / 256), 256, 0, stream>>>(x, (float*)d_out);
}
